// HNLQLinearQAT_28656021799227
// MI455X (gfx1250) — 8-bit matrix operand order repaired and checked by reading, compiled, not run on hardware
//
#include <hip/hip_runtime.h>
#include <cstddef>
#include <cstdint>

typedef __attribute__((ext_vector_type(8))) int v8i;

#define N_DIM 4096
#define K_DIM 4096

__global__ __launch_bounds__(256)
void lsq_act_quant(const float* __restrict__ x,
                   const float* __restrict__ alpha,
                   char* __restrict__ kq) {
    const size_t i = ((size_t)blockIdx.x * blockDim.x + threadIdx.x) * 4;
    const float a   = fmaxf(alpha[0], 0.0f) + 1e-8f;
    const float inv = 127.0f / a;
    const float4 xv = *reinterpret_cast<const float4*>(x + i);
    const int b0 = __float2int_rn(fminf(fmaxf(xv.x, -a), a) * inv);
    const int b1 = __float2int_rn(fminf(fmaxf(xv.y, -a), a) * inv);
    const int b2 = __float2int_rn(fminf(fmaxf(xv.z, -a), a) * inv);
    const int b3 = __float2int_rn(fminf(fmaxf(xv.w, -a), a) * inv);
    const int packed = (b0 & 0xff) | ((b1 & 0xff) << 8) |
                       ((b2 & 0xff) << 16) | ((b3 & 0xff) << 24);
    *reinterpret_cast<int*>(kq + i) = packed;
}

__global__ __launch_bounds__(256)
void hnlq_weight_quant(const float* __restrict__ W,
                       const float* __restrict__ theta_beta,
                       const float* __restrict__ sigma_ema,
                       const float* __restrict__ G,
                       const float* __restrict__ Gi,
                       char* __restrict__ hq,
                       float* __restrict__ wscale) {
    __shared__ float sG[64];
    __shared__ float sGi[64];
    const int tid = threadIdx.x;
    if (tid < 64)       sG[tid] = G[tid];
    else if (tid < 128) sGi[tid - 64] = Gi[tid - 64];
    __syncthreads();

    const size_t gid = (size_t)blockIdx.x * blockDim.x + tid;
    const int o = (int)(gid >> 9);
    const float Delta = 0.05f * expf(theta_beta[o]) * sigma_ema[o];
    const float invD  = 1.0f / Delta;

    float r[8], xhat[8];
    const float* wp = W + gid * 8;
#pragma unroll
    for (int d = 0; d < 8; ++d) {
        const float wc = fminf(fmaxf(wp[d], -2.0f), 2.0f);
        r[d] = wc * invD;
        xhat[d] = 0.0f;
    }
#pragma unroll
    for (int m = 1; m >= 0; --m) {
        const float s  = (m == 1) ? 4.0f : 1.0f;
        const float is = (m == 1) ? 0.25f : 1.0f;
        float z[8];
#pragma unroll
        for (int k = 0; k < 8; ++k) {
            float acc = 0.0f;
#pragma unroll
            for (int d = 0; d < 8; ++d) acc += r[d] * sGi[k * 8 + d];
            const float zr = rintf(acc * is);
            z[k] = fminf(fmaxf(zr, -2.0f), 2.0f);
        }
#pragma unroll
        for (int d = 0; d < 8; ++d) {
            float qm = 0.0f;
#pragma unroll
            for (int k = 0; k < 8; ++k) qm += z[k] * sG[k * 8 + d];
            qm *= s;
            xhat[d] += qm;
            r[d]    -= qm;
        }
    }
    unsigned long long pk = 0ull;
#pragma unroll
    for (int d = 0; d < 8; ++d) {
        const int hv = __float2int_rn(2.0f * xhat[d]);
        pk |= ((unsigned long long)(hv & 0xff)) << (8 * d);
    }
    *reinterpret_cast<unsigned long long*>(hq + gid * 8) = pk;
    if ((gid & 511) == 0) wscale[o] = 0.5f * Delta;
}

__device__ __forceinline__ void async_ld16(unsigned long long gbase,
                                           uint32_t goff, uint32_t ldsaddr) {
    asm volatile("global_load_async_to_lds_b128 %0, %1, %2"
                 :: "v"(ldsaddr), "v"(goff), "s"(gbase)
                 : "memory");
}

__global__ __launch_bounds__(256)
void iu8_wmma_gemm(const char* __restrict__ Aq,
                   const char* __restrict__ Hq,
                   const float* __restrict__ wscale,
                   const float* __restrict__ bias,
                   const float* __restrict__ alpha,
                   float* __restrict__ out) {
    constexpr int KSTEP = 128;
    constexpr int LDT   = 144;
    constexpr int BOFF  = 128 * LDT;
    constexpr int BUFSZ = 2 * 128 * LDT;
    __shared__ __align__(16) char smem[2 * BUFSZ];

    const int tid  = threadIdx.x;
    const int lane = tid & 31;
    const int wave = tid >> 5;
    const int wm   = wave & 3;
    const int wn   = wave >> 2;
    const int mBase = blockIdx.y * 128;
    const int nBase = blockIdx.x * 128;

    const int g  = lane >> 4;
    const int lr = lane & 15;

    const unsigned long long aBase =
        (unsigned long long)(uintptr_t)(Aq + (size_t)mBase * K_DIM);
    const unsigned long long bBase =
        (unsigned long long)(uintptr_t)(Hq + (size_t)nBase * K_DIM);
    const uint32_t smemBase = (uint32_t)(uintptr_t)(void*)smem;

    v8i acc[2][4];
#pragma unroll
    for (int i = 0; i < 2; ++i)
#pragma unroll
        for (int j = 0; j < 4; ++j) acc[i][j] = (v8i)0;

    union FA { unsigned long long u[4]; v8i v; };
    union FB { int4 q[2]; v8i v; };

    auto stage = [&](uint32_t k0, uint32_t ldsbuf) {
#pragma unroll
        for (int it = 0; it < 4; ++it) {
            const uint32_t c   = (uint32_t)tid + 256u * it;
            const uint32_t row = c >> 3;
            const uint32_t ko  = (c & 7) * 16;
            async_ld16(aBase, row * K_DIM + k0 + ko, ldsbuf + row * LDT + ko);
            async_ld16(bBase, row * K_DIM + k0 + ko, ldsbuf + BOFF + row * LDT + ko);
        }
    };

    stage(0, smemBase);

    int buf = 0;
    for (int k0 = 0; k0 < K_DIM; k0 += KSTEP) {
        if (k0 + KSTEP < K_DIM) {
            stage((uint32_t)k0 + KSTEP, smemBase + (uint32_t)(buf ^ 1) * BUFSZ);
            asm volatile("s_wait_asynccnt 8" ::: "memory");
        } else {
            asm volatile("s_wait_asynccnt 0" ::: "memory");
        }
        __syncthreads();

        const char* lA = &smem[buf * BUFSZ];
        const char* lB = &smem[buf * BUFSZ + BOFF];

#pragma unroll
        for (int s = 0; s < 2; ++s) {
            const int kb = s * 64;
            FA fa[2];
#pragma unroll
            for (int i = 0; i < 2; ++i) {
                const char* base = &lA[(wm * 32 + i * 16 + lr) * LDT + kb + g * 16];
                fa[i].u[0] = *reinterpret_cast<const unsigned long long*>(base + 0);
                fa[i].u[1] = *reinterpret_cast<const unsigned long long*>(base + 8);
                fa[i].u[2] = *reinterpret_cast<const unsigned long long*>(base + 32);
                fa[i].u[3] = *reinterpret_cast<const unsigned long long*>(base + 40);
            }
            FB fb[4];
#pragma unroll
            for (int j = 0; j < 4; ++j) {
                const char* base = &lB[(wn * 64 + j * 16 + lr) * LDT + kb + g * 16];
                fb[j].q[0] = *reinterpret_cast<const int4*>(base + 0);
                fb[j].q[1] = *reinterpret_cast<const int4*>(base + 32);
            }
#pragma unroll
            for (int i = 0; i < 2; ++i)
#pragma unroll
                for (int j = 0; j < 4; ++j)
                    acc[i][j] = __builtin_amdgcn_wmma_i32_16x16x64_iu8(
                         true, fa[i].v,  true, fb[j].v,
                        acc[i][j],  false,  false);
        }
        __syncthreads();
        buf ^= 1;
    }

    const float a      = fmaxf(alpha[0], 0.0f) + 1e-8f;
    const float ascale = a / 127.0f;
#pragma unroll
    for (int i = 0; i < 2; ++i) {
        const int rowBase = mBase + wm * 32 + i * 16 + 8 * g;
#pragma unroll
        for (int j = 0; j < 4; ++j) {
            const int col = nBase + wn * 64 + j * 16 + lr;
            const float f = ascale * wscale[col];
            const float b = bias[col];
#pragma unroll
            for (int v = 0; v < 8; ++v) {
                out[(size_t)(rowBase + v) * N_DIM + col] =
                    (float)acc[i][j][v] * f + b;
            }
        }
    }
}

extern "C" void kernel_launch(void* const* d_in, const int* in_sizes, int n_in,
                              void* d_out, int out_size, void* d_ws, size_t ws_size,
                              hipStream_t stream) {
    (void)in_sizes; (void)n_in; (void)out_size; (void)ws_size;
    const float* x      = (const float*)d_in[0];
    const float* weight = (const float*)d_in[1];
    const float* bias   = (const float*)d_in[2];
    const float* theta  = (const float*)d_in[3];
    const float* alpha  = (const float*)d_in[4];
    const float* sigma  = (const float*)d_in[5];
    const float* G      = (const float*)d_in[6];
    const float* Gi     = (const float*)d_in[7];
    float* out = (float*)d_out;

    const size_t MAT = (size_t)N_DIM * K_DIM;
    char*  Kq = (char*)d_ws;
    char*  Hq = Kq + MAT;
    float* ws = (float*)(Hq + MAT);

    lsq_act_quant<<<(unsigned)(MAT / (256 * 4)), 256, 0, stream>>>(x, alpha, Kq);
    hnlq_weight_quant<<<(unsigned)(MAT / 8 / 256), 256, 0, stream>>>(
        weight, theta, sigma, G, Gi, Hq, ws);
    dim3 grid(N_DIM / 128, N_DIM / 128);
    iu8_wmma_gemm<<<grid, 256, 0, stream>>>(Kq, Hq, ws, bias, alpha, out);
}
